// ResGCN_30202210025889
// MI455X (gfx1250) — matrix operand order repaired and checked by reading, not compiled, not run on hardware
//
#include <hip/hip_runtime.h>
#include <hip/hip_bf16.h>

#define N_H 128
#define N_C 10
#define N_G 64

typedef __attribute__((ext_vector_type(16))) __bf16          v16bf;
typedef __attribute__((ext_vector_type(16))) unsigned short  v16us;
typedef __attribute__((ext_vector_type(8)))  unsigned short  v8us;
typedef __attribute__((ext_vector_type(8)))  float           v8f;

__device__ __forceinline__ unsigned short f2bf(float f) {
    unsigned int u = __float_as_uint(f);
    unsigned int r = u + 0x7FFFu + ((u >> 16) & 1u);   // round-to-nearest-even
    return (unsigned short)(r >> 16);
}

// ---------------- small utility kernels ----------------

__global__ void zero_f32_kernel(float* p, int n) {
    int i = blockIdx.x * blockDim.x + threadIdx.x;
    if (i < n) p[i] = 0.0f;
}

__global__ void deg_kernel(const int* __restrict__ dst, float* deg, int E) {
    int e = blockIdx.x * blockDim.x + threadIdx.x;
    if (e < E) atomicAdd(&deg[dst[e]], 1.0f);
}

__global__ void dinv_kernel(float* deg, int n) {
    int i = blockIdx.x * blockDim.x + threadIdx.x;
    if (i < n) deg[i] = rsqrtf(deg[i] + 1.0f);   // +1 self loop; deg >= 1 always
}

__global__ void bias_fill_kernel(float* __restrict__ out, const float* __restrict__ b, int n) {
    int idx = blockIdx.x * blockDim.x + threadIdx.x;
    if (idx < n * N_H) out[idx] = b[idx & (N_H - 1)];
}

__global__ void f32_to_bf16_kernel(const float* __restrict__ in, unsigned short* __restrict__ out, int n) {
    int idx = blockIdx.x * blockDim.x + threadIdx.x;
    if (idx < n) out[idx] = f2bf(in[idx]);
}

// Pack fp32 weight W[k][n] (128x128 row-major) into bf16 B-operand layout:
// packed[((tn*4+kk)*32+lane)*16+j] = bf16(W[kk*32 + (lane>>4)*16 + j][tn*16 + (lane&15)])
__global__ void pack_w_kernel(const float* __restrict__ W, unsigned short* __restrict__ Wp) {
    int idx = blockIdx.x * blockDim.x + threadIdx.x;
    if (idx >= 8 * 4 * 32 * 16) return;
    int j    = idx & 15;
    int lane = (idx >> 4) & 31;
    int kk   = (idx >> 9) & 3;
    int tn   = (idx >> 11);
    int k = kk * 32 + ((lane >> 4) * 16) + j;
    int n = tn * 16 + (lane & 15);
    Wp[idx] = f2bf(W[k * N_H + n]);
}

// ---------------- WMMA GEMM: out[M x 128] = bf16in[M x 128] @ W + (bias) ----------------
// One wave per 16-row strip; 8 WMMA column tiles; K loop of 4 x 32 (bf16).
// A operand already bf16 in memory: pure loads + WMMA in the hot loop.

struct APair { v8us lo, hi; };

__global__ __launch_bounds__(128) void gemm_wmma_kernel(
    const unsigned short* __restrict__ inbf, const unsigned short* __restrict__ Wp,
    const float* __restrict__ bias, float* __restrict__ out, int M)
{
    int wave = blockIdx.x * (blockDim.x >> 5) + (threadIdx.x >> 5);
    int lane = threadIdx.x & 31;
    int nstrips = M >> 4;
    if (wave >= nstrips) return;          // wave-uniform: EXEC all-ones inside WMMA

    int row0 = wave << 4;
    int arow = row0 + (lane & 15);
    int agrp = lane >> 4;                 // 0 or 1 (K-half group)
    const unsigned short* ap = inbf + (size_t)arow * N_H;

    v8f acc[8];
#pragma unroll
    for (int t = 0; t < 8; ++t) acc[t] = (v8f){0.f,0.f,0.f,0.f,0.f,0.f,0.f,0.f};

#pragma unroll
    for (int kk = 0; kk < 4; ++kk) {
        int kbase = kk * 32 + agrp * 16;
        // halves 0..7 -> K kbase..kbase+7 ; halves 8..15 -> K kbase+8..kbase+15
        APair pr;
        pr.lo = *reinterpret_cast<const v8us*>(ap + kbase);
        pr.hi = *reinterpret_cast<const v8us*>(ap + kbase + 8);
        v16bf a = __builtin_bit_cast(v16bf, pr);

#pragma unroll
        for (int t = 0; t < 8; ++t) {
            const v16us* bp = reinterpret_cast<const v16us*>(Wp) + ((t * 4 + kk) * 32 + lane);
            v16bf b = __builtin_bit_cast(v16bf, *bp);
            acc[t] = __builtin_amdgcn_wmma_f32_16x16x32_bf16(
                false, a, false, b, (short)0, acc[t], false, false);
        }
    }

    // D layout: VGPR v, lane l -> M = v + (l>>4)*8, N = (l&15)
    int col  = lane & 15;
    int mofs = (lane >> 4) * 8;
#pragma unroll
    for (int t = 0; t < 8; ++t) {
        float bv = bias ? bias[t * 16 + col] : 0.0f;
#pragma unroll
        for (int v = 0; v < 8; ++v) {
            int r = row0 + mofs + v;
            out[(size_t)r * N_H + t * 16 + col] = acc[t][v] + bv;
        }
    }
}

// ---------------- SpMM: resbase[dst] += (dinv[src]*dinv[dst]) * lin[src] ----------------

__global__ void spmm_kernel(const float* __restrict__ lin, const int* __restrict__ src,
                            const int* __restrict__ dst, const float* __restrict__ dinv,
                            float* outb, int E)
{
    long long gid = (long long)blockIdx.x * blockDim.x + threadIdx.x;
    int e = (int)(gid >> 5);
    if (e >= E) return;
    int c = ((int)gid & 31) * 4;
    int s = src[e], d = dst[e];
    float coef = dinv[s] * dinv[d];
    float4 m = *reinterpret_cast<const float4*>(lin + (size_t)s * N_H + c);
    float* o = outb + (size_t)d * N_H + c;
    atomicAdd(o + 0, coef * m.x);
    atomicAdd(o + 1, coef * m.y);
    atomicAdd(o + 2, coef * m.z);
    atomicAdd(o + 3, coef * m.w);
}

// ---------------- BN(eval) + ReLU, self-loop folded in; dual fp32 + bf16 output ----------------

__global__ void bn_relu_kernel(const float* __restrict__ resbase, const float* lin,
                               const float* __restrict__ dinv,
                               const float* __restrict__ gamma, const float* __restrict__ beta,
                               float* outf, unsigned short* __restrict__ outbf, int n)
{
    int idx = blockIdx.x * blockDim.x + threadIdx.x;
    if (idx >= n * N_H) return;
    int i = idx >> 7, f = idx & (N_H - 1);
    float dv = dinv[i];
    float val = resbase[idx] + dv * dv * lin[idx];     // + self-loop message
    val = val * 0.99999499998750f;                     // rsqrt(1 + 1e-5)
    val = val * gamma[f] + beta[f];
    val = fmaxf(val, 0.0f);
    outf[idx]  = val;
    outbf[idx] = f2bf(val);
}

// ---------------- global max pool (values >= 0 -> int-bits atomicMax) ----------------

__global__ void pool_kernel(const float* __restrict__ h, const int* __restrict__ batch,
                            int* pooled, int n)
{
    int idx = blockIdx.x * blockDim.x + threadIdx.x;
    if (idx >= n * N_H) return;
    int i = idx >> 7, f = idx & (N_H - 1);
    atomicMax(&pooled[batch[i] * N_H + f], __float_as_int(h[idx]));
}

// ---------------- MLP head (tiny, scalar) ----------------

__global__ void head1_kernel(const float* __restrict__ pooled, const float* __restrict__ W,
                             const float* __restrict__ b, float* __restrict__ z1)
{
    int idx = blockIdx.x * blockDim.x + threadIdx.x;
    if (idx >= N_G * N_H) return;
    int g = idx >> 7, n = idx & (N_H - 1);
    float s = b[n];
    for (int k = 0; k < N_H; ++k) s += pooled[g * N_H + k] * W[k * N_H + n];
    z1[idx] = fmaxf(s, 0.0f);
}

__global__ void head2_kernel(const float* __restrict__ z1, const float* __restrict__ W,
                             const float* __restrict__ b, float* __restrict__ out)
{
    int idx = blockIdx.x * blockDim.x + threadIdx.x;
    if (idx >= N_G * N_C) return;
    int g = idx / N_C, c = idx % N_C;
    float s = b[c];
    for (int k = 0; k < N_H; ++k) s += z1[g * N_H + k] * W[k * N_C + c];
    out[idx] = s;
}

// ---------------- launcher ----------------

extern "C" void kernel_launch(void* const* d_in, const int* in_sizes, int n_in,
                              void* d_out, int out_size, void* d_ws, size_t ws_size,
                              hipStream_t stream) {
    const float* x    = (const float*)d_in[0];
    const int*   ei   = (const int*)d_in[1];
    const int*   batch= (const int*)d_in[2];
    const float* W0   = (const float*)d_in[3];
    const float* b0   = (const float*)d_in[4];
    const float* g0   = (const float*)d_in[5];
    const float* be0  = (const float*)d_in[6];
    const float* Wh   = (const float*)d_in[7];
    const float* bh   = (const float*)d_in[8];
    const float* gh   = (const float*)d_in[9];
    const float* beh  = (const float*)d_in[10];
    const float* Rh   = (const float*)d_in[11];
    const float* mW1  = (const float*)d_in[12];
    const float* mb1  = (const float*)d_in[13];
    const float* mW2  = (const float*)d_in[14];
    const float* mb2  = (const float*)d_in[15];
    float* out = (float*)d_out;

    const int N = in_sizes[2];          // 50000
    const int E = in_sizes[1] / 2;      // 800000
    const int* src = ei;
    const int* dst = ei + E;

    // workspace layout (256B aligned slabs)
    char* ws = (char*)d_ws;
    size_t o = 0;
    auto slab = [&](size_t bytes) { char* p = ws + o; o += (bytes + 255) & ~(size_t)255; return p; };
    float*          dinv  = (float*)slab((size_t)N * 4);
    float*          bufL  = (float*)slab((size_t)N * N_H * 4);    // lin / fp32 h
    float*          bufR  = (float*)slab((size_t)N * N_H * 4);    // resbase
    unsigned short* hbf   = (unsigned short*)slab((size_t)N * N_H * 2);  // bf16 activations (A operand)
    unsigned short* packW = (unsigned short*)slab(7 * 8 * 4 * 32 * 16 * 2);
    float*          pooled= (float*)slab(N_G * N_H * 4);
    float*          z1    = (float*)slab(N_G * N_H * 4);

    const int T = 256;
    const int NH_blocks = (N * N_H + T - 1) / T;

    // degree -> dinv
    zero_f32_kernel<<<(N + T - 1) / T, T, 0, stream>>>(dinv, N);
    deg_kernel<<<(E + T - 1) / T, T, 0, stream>>>(dst, dinv, E);
    dinv_kernel<<<(N + T - 1) / T, T, 0, stream>>>(dinv, N);

    // pack weights to bf16 WMMA-B layout: [W0, Wh0..2, Rh0..2]
    unsigned short* pW0 = packW;
    unsigned short* pWh[3], *pRh[3];
    for (int i = 0; i < 3; ++i) { pWh[i] = packW + (1 + i) * 16384; pRh[i] = packW + (4 + i) * 16384; }
    pack_w_kernel<<<64, T, 0, stream>>>(W0, pW0);
    for (int i = 0; i < 3; ++i) {
        pack_w_kernel<<<64, T, 0, stream>>>(Wh + (size_t)i * N_H * N_H, pWh[i]);
        pack_w_kernel<<<64, T, 0, stream>>>(Rh + (size_t)i * N_H * N_H, pRh[i]);
    }

    const int gemm_blocks = ((N >> 4) + 3) / 4;   // 4 waves / 128-thread block

    // ---- layer 0: h = bn_relu(spmm(x@W0) + b0) ----
    f32_to_bf16_kernel<<<NH_blocks, T, 0, stream>>>(x, hbf, N * N_H);
    gemm_wmma_kernel<<<gemm_blocks, 128, 0, stream>>>(hbf, pW0, nullptr, bufL, N);
    bias_fill_kernel<<<NH_blocks, T, 0, stream>>>(bufR, b0, N);
    spmm_kernel<<<(int)(((long long)E * 32 + T - 1) / T), T, 0, stream>>>(bufL, src, dst, dinv, bufR, E);
    bn_relu_kernel<<<NH_blocks, T, 0, stream>>>(bufR, bufL, dinv, g0, be0, bufL, hbf, N);

    // ---- residual layers: c = spmm(h@Wh) + bh + h@Rh ; h = bn_relu(c) ----
    for (int i = 0; i < 3; ++i) {
        gemm_wmma_kernel<<<gemm_blocks, 128, 0, stream>>>(hbf, pWh[i], nullptr, bufL, N);
        gemm_wmma_kernel<<<gemm_blocks, 128, 0, stream>>>(hbf, pRh[i], bh + i * N_H, bufR, N);
        spmm_kernel<<<(int)(((long long)E * 32 + T - 1) / T), T, 0, stream>>>(bufL, src, dst, dinv, bufR, E);
        bn_relu_kernel<<<NH_blocks, T, 0, stream>>>(bufR, bufL, dinv, gh + i * N_H, beh + i * N_H, bufL, hbf, N);
    }

    // ---- pooling + head (final fp32 h lives in bufL) ----
    zero_f32_kernel<<<(N_G * N_H + T - 1) / T, T, 0, stream>>>(pooled, N_G * N_H);
    pool_kernel<<<NH_blocks, T, 0, stream>>>(bufL, batch, (int*)pooled, N);
    head1_kernel<<<(N_G * N_H + T - 1) / T, T, 0, stream>>>(pooled, mW1, mb1, z1);
    head2_kernel<<<(N_G * N_C + T - 1) / T, T, 0, stream>>>(z1, mW2, mb2, out);
}
